// NetGRU_68942815036006
// MI455X (gfx1250) — matrix operand order repaired and checked by reading, not compiled, not run on hardware
//
#include <hip/hip_runtime.h>
#include <hip/hip_bf16.h>
#include <math.h>

typedef __bf16 bf16_t;
typedef __attribute__((ext_vector_type(16))) __bf16 v16bf;
typedef __attribute__((ext_vector_type(8)))  float  v8f;

#define HID   1024
#define BATCH 64
#define TLEN  512
#define INDIM 118

// ---------------------------------------------------------------------------
// A-fragment lane mapping for 16-bit A 16x32 (M x K), wave32 (ISA 7.12.2):
//   lanes 0-15 : M=lane,    e(0..7)->K=e,      e(8..15)->K=e+8   (K 0-7,16-23)
//   lanes16-31 : M=lane-16, e(0..7)->K=e+8,    e(8..15)->K=e+16  (K 8-15,24-31)
// B-fragment (32x16 K x N): lanes 0-15: N=lane ; lanes16-31: N=lane-16 ; K=a_klocal(lane,e), same as A
// C/D f32 16x16: vgpr r: lanes0-15 -> M=r, N=lane ; lanes16-31 -> M=8+r, N=lane-16
// ---------------------------------------------------------------------------

__device__ __forceinline__ int a_klocal(int lane, int e) {
    if (lane < 16) return (e < 8) ? e : e + 8;
    else           return (e < 8) ? e + 8 : e + 16;
}

// Pack W (3H x Kact row-major, fp32) into bf16 B-fragments:
// dst[((ntile*KT + kt)*32 + lane)*16 + e], n = ntile*16 + lane%16,
// k = kt*32 + a_klocal(lane, e) ; zero-pad k >= Kact.
__global__ void pack_b_kernel(const float* __restrict__ W, int Kact, int KT,
                              bf16_t* __restrict__ dst) {
    int idx = blockIdx.x * blockDim.x + threadIdx.x;
    int total = 192 * KT * 512;               // 192 n-tiles (3H/16)
    if (idx >= total) return;
    int e     = idx & 15;
    int lane  = (idx >> 4) & 31;
    int kt    = (idx >> 9) % KT;
    int ntile = idx / (512 * KT);
    int n = ntile * 16 + (lane & 15);
    int k = kt * 32 + a_klocal(lane, e);
    float v = (k < Kact) ? W[n * Kact + k] : 0.0f;
    dst[idx] = (bf16_t)v;
}

// Pack x (B,T,I fp32) into per-timestep A-fragments, K padded 118->128.
// dst layout: [t][kt(4)][mt(4)][lane(32)][e(16)]
__global__ void pack_x_kernel(const float* __restrict__ x, bf16_t* __restrict__ dst) {
    int idx = blockIdx.x * blockDim.x + threadIdx.x;   // total 512*8192
    if (idx >= TLEN * 8192) return;
    int e    = idx & 15;
    int lane = (idx >> 4) & 31;
    int mt   = (idx >> 9) & 3;
    int kt   = (idx >> 11) & 3;
    int t    = idx >> 13;
    int m = mt * 16 + (lane & 15);
    int k = kt * 32 + a_klocal(lane, e);
    float v = (k < INDIM) ? x[(m * TLEN + t) * INDIM + k] : 0.0f;
    dst[idx] = (bf16_t)v;
}

__global__ void zero_kernel(unsigned int* __restrict__ p, int n) {
    int idx = blockIdx.x * blockDim.x + threadIdx.x;
    if (idx < n) p[idx] = 0u;
}

// ---------------------------------------------------------------------------
// Fused GRU cell: gi = Ai @ Bih + bih ; gh = Ah @ Bhh + bhh ; GRU nonlinearity.
// grid.x = 64 (16-column chunks of H), block = 128 (4 waves, one M-tile each).
// Epilogue writes fp32 h_new and repacks the chunk into bf16 A-fragments.
// ---------------------------------------------------------------------------
__global__ void __launch_bounds__(128)
gru_cell_kernel(const bf16_t* __restrict__ Ai,  const bf16_t* __restrict__ Bih,
                const float*  __restrict__ bih, int KTi,
                const bf16_t* __restrict__ Ah,  const bf16_t* __restrict__ Bhh,
                const float*  __restrict__ bhh,
                const float*  __restrict__ hprev, float* __restrict__ hnew,
                bf16_t* __restrict__ packA_out) {
    const int lane = threadIdx.x & 31;
    const int wave = threadIdx.x >> 5;     // M-tile 0..3
    const int c    = blockIdx.x;           // H-column chunk 0..63

    __shared__ float smem[BATCH * 16];

    v8f accI0 = {}, accI1 = {}, accI2 = {};
    v8f accH0 = {}, accH1 = {}, accH2 = {};

    // ---- input-gate GEMM: gi = Ai @ Wih^T, K = KTi*32 ----
    for (int kt = 0; kt < KTi; ++kt) {
        v16bf a = *(const v16bf*)(Ai + (((kt << 2) + wave) * 32 + lane) * 16);
        const bf16_t* b0 = Bih + (((size_t)(0 * 64 + c) * KTi + kt) * 512) + lane * 16;
        const bf16_t* b1 = Bih + (((size_t)(1 * 64 + c) * KTi + kt) * 512) + lane * 16;
        const bf16_t* b2 = Bih + (((size_t)(2 * 64 + c) * KTi + kt) * 512) + lane * 16;
        __builtin_prefetch(b0 + 512, 0, 1);
        accI0 = __builtin_amdgcn_wmma_f32_16x16x32_bf16(false, a, false, *(const v16bf*)b0,
                                                        (short)0, accI0, false, false);
        accI1 = __builtin_amdgcn_wmma_f32_16x16x32_bf16(false, a, false, *(const v16bf*)b1,
                                                        (short)0, accI1, false, false);
        accI2 = __builtin_amdgcn_wmma_f32_16x16x32_bf16(false, a, false, *(const v16bf*)b2,
                                                        (short)0, accI2, false, false);
    }

    // ---- hidden-gate GEMM: gh = Ah @ Whh^T, K = 1024 (32 k-tiles) ----
    for (int kt = 0; kt < 32; ++kt) {
        v16bf a = *(const v16bf*)(Ah + (((kt << 2) + wave) * 32 + lane) * 16);
        const bf16_t* b0 = Bhh + (((size_t)(0 * 64 + c) * 32 + kt) * 512) + lane * 16;
        const bf16_t* b1 = Bhh + (((size_t)(1 * 64 + c) * 32 + kt) * 512) + lane * 16;
        const bf16_t* b2 = Bhh + (((size_t)(2 * 64 + c) * 32 + kt) * 512) + lane * 16;
        __builtin_prefetch(b0 + 512, 0, 1);
        accH0 = __builtin_amdgcn_wmma_f32_16x16x32_bf16(false, a, false, *(const v16bf*)b0,
                                                        (short)0, accH0, false, false);
        accH1 = __builtin_amdgcn_wmma_f32_16x16x32_bf16(false, a, false, *(const v16bf*)b1,
                                                        (short)0, accH1, false, false);
        accH2 = __builtin_amdgcn_wmma_f32_16x16x32_bf16(false, a, false, *(const v16bf*)b2,
                                                        (short)0, accH2, false, false);
    }

    // ---- fused GRU update ----
    const int nloc = lane & 15;
    const int col  = c * 16 + nloc;
    const float br = bih[col],        bz = bih[HID + col],     bn = bih[2 * HID + col];
    const float cr = bhh[col],        cz = bhh[HID + col],     cn = bhh[2 * HID + col];
    #pragma unroll
    for (int r = 0; r < 8; ++r) {
        int mloc = r + ((lane >= 16) ? 8 : 0);
        int m    = wave * 16 + mloc;
        float ir  = accI0[r] + br;
        float iz  = accI1[r] + bz;
        float inn = accI2[r] + bn;
        float hr  = accH0[r] + cr;
        float hz  = accH1[r] + cz;
        float hn  = accH2[r] + cn;
        float rg = 1.0f / (1.0f + __expf(-(ir + hr)));
        float zg = 1.0f / (1.0f + __expf(-(iz + hz)));
        float ng = tanhf(inn + rg * hn);
        float hp = hprev[m * HID + col];
        float hv = (1.0f - zg) * ng + zg * hp;
        hnew[m * HID + col] = hv;
        smem[m * 16 + nloc] = hv;
    }
    __syncthreads();

    // ---- repack this chunk of h_new into bf16 A-fragments for the next GEMM ----
    #pragma unroll
    for (int p = 0; p < 8; ++p) {
        int idx = threadIdx.x * 8 + p;           // 0..1023 over 64x16 chunk
        int m  = idx >> 4;
        int nl = idx & 15;
        float v = smem[idx];
        int k  = c * 16 + nl;
        int kt = k >> 5, kk = k & 31, mt = m >> 4, mr = m & 15;
        int lp, e;
        if      (kk < 8)  { lp = mr;      e = kk;      }
        else if (kk < 16) { lp = mr + 16; e = kk - 8;  }
        else if (kk < 24) { lp = mr;      e = kk - 8;  }
        else              { lp = mr + 16; e = kk - 16; }
        packA_out[(((kt << 2) + mt) * 32 + lp) * 16 + e] = (bf16_t)v;
    }
}

// ---------------------------------------------------------------------------
// y = relu(h2 @ fc_W^T + fc_b) @ out_W^T + out_b   (single small block)
// ---------------------------------------------------------------------------
__global__ void __launch_bounds__(256)
fc_head_kernel(const float* __restrict__ h2, const float* __restrict__ fcW,
               const float* __restrict__ fcb, const float* __restrict__ outW,
               const float* __restrict__ outb, float* __restrict__ out) {
    __shared__ float y1[BATCH * 16];
    int tid = threadIdx.x;
    #pragma unroll
    for (int p = 0; p < 4; ++p) {
        int idx = tid + p * 256;       // 0..1023
        int m = idx >> 4, j = idx & 15;
        float s = fcb[j];
        for (int k = 0; k < HID; ++k) s += h2[m * HID + k] * fcW[j * HID + k];
        y1[idx] = fmaxf(s, 0.0f);
    }
    __syncthreads();
    if (tid < BATCH) {
        float s = outb[0];
        #pragma unroll
        for (int j = 0; j < 16; ++j) s += y1[tid * 16 + j] * outW[j];
        out[tid] = s;
    }
}

extern "C" void kernel_launch(void* const* d_in, const int* in_sizes, int n_in,
                              void* d_out, int out_size, void* d_ws, size_t ws_size,
                              hipStream_t stream) {
    const float* x      = (const float*)d_in[0];
    const float* eWih0  = (const float*)d_in[1];
    const float* eWhh0  = (const float*)d_in[2];
    const float* ebih0  = (const float*)d_in[3];
    const float* ebhh0  = (const float*)d_in[4];
    const float* eWih1  = (const float*)d_in[5];
    const float* eWhh1  = (const float*)d_in[6];
    const float* ebih1  = (const float*)d_in[7];
    const float* ebhh1  = (const float*)d_in[8];
    const float* dWih0  = (const float*)d_in[9];
    const float* dWhh0  = (const float*)d_in[10];
    const float* dbih0  = (const float*)d_in[11];
    const float* dbhh0  = (const float*)d_in[12];
    const float* dWih1  = (const float*)d_in[13];
    const float* dWhh1  = (const float*)d_in[14];
    const float* dbih1  = (const float*)d_in[15];
    const float* dbhh1  = (const float*)d_in[16];
    const float* fcW    = (const float*)d_in[17];
    const float* fcb    = (const float*)d_in[18];
    const float* outW   = (const float*)d_in[19];
    const float* outb   = (const float*)d_in[20];
    float* out = (float*)d_out;

    char* ws = (char*)d_ws;
    size_t o = 0;
    auto suballoc = [&](size_t bytes) { size_t r = o; o = (o + bytes + 255) & ~(size_t)255; return r; };
    const size_t PB_SMALL = 192UL * 4  * 512 * 2;   // 3072 x 128 bf16 fragments
    const size_t PB_BIG   = 192UL * 32 * 512 * 2;   // 3072 x 1024 bf16 fragments
    size_t oPBeWih0 = suballoc(PB_SMALL);
    size_t oPBeWhh0 = suballoc(PB_BIG);
    size_t oPBeWih1 = suballoc(PB_BIG);
    size_t oPBeWhh1 = suballoc(PB_BIG);
    size_t oPBdWih0 = suballoc(PB_SMALL);
    size_t oPBdWhh0 = suballoc(PB_BIG);
    size_t oPBdWih1 = suballoc(PB_BIG);
    size_t oPBdWhh1 = suballoc(PB_BIG);
    size_t oPAX     = suballoc((size_t)TLEN * 8192 * 2);      // per-step x fragments
    size_t oH1      = suballoc(2UL * BATCH * HID * 4);        // fp32 ping-pong
    size_t oH2      = suballoc(2UL * BATCH * HID * 4);
    size_t oPAh1    = suballoc(2UL * BATCH * HID * 2);        // bf16 fragment ping-pong
    size_t oPAh2    = suballoc(2UL * BATCH * HID * 2);

    bf16_t* PBeWih0 = (bf16_t*)(ws + oPBeWih0);
    bf16_t* PBeWhh0 = (bf16_t*)(ws + oPBeWhh0);
    bf16_t* PBeWih1 = (bf16_t*)(ws + oPBeWih1);
    bf16_t* PBeWhh1 = (bf16_t*)(ws + oPBeWhh1);
    bf16_t* PBdWih0 = (bf16_t*)(ws + oPBdWih0);
    bf16_t* PBdWhh0 = (bf16_t*)(ws + oPBdWhh0);
    bf16_t* PBdWih1 = (bf16_t*)(ws + oPBdWih1);
    bf16_t* PBdWhh1 = (bf16_t*)(ws + oPBdWhh1);
    bf16_t* PAX     = (bf16_t*)(ws + oPAX);
    float*  H1      = (float*)(ws + oH1);
    float*  H2      = (float*)(ws + oH2);
    bf16_t* PAh1    = (bf16_t*)(ws + oPAh1);
    bf16_t* PAh2    = (bf16_t*)(ws + oPAh2);

    // ---- one-time packing (re-done every call for determinism) ----
    const int nSmall = 192 * 4 * 512, nBig = 192 * 32 * 512;
    pack_b_kernel<<<(nSmall + 255) / 256, 256, 0, stream>>>(eWih0, INDIM, 4,  PBeWih0);
    pack_b_kernel<<<(nBig   + 255) / 256, 256, 0, stream>>>(eWhh0, HID,   32, PBeWhh0);
    pack_b_kernel<<<(nBig   + 255) / 256, 256, 0, stream>>>(eWih1, HID,   32, PBeWih1);
    pack_b_kernel<<<(nBig   + 255) / 256, 256, 0, stream>>>(eWhh1, HID,   32, PBeWhh1);
    pack_b_kernel<<<(nSmall + 255) / 256, 256, 0, stream>>>(dWih0, INDIM, 4,  PBdWih0);
    pack_b_kernel<<<(nBig   + 255) / 256, 256, 0, stream>>>(dWhh0, HID,   32, PBdWhh0);
    pack_b_kernel<<<(nBig   + 255) / 256, 256, 0, stream>>>(dWih1, HID,   32, PBdWih1);
    pack_b_kernel<<<(nBig   + 255) / 256, 256, 0, stream>>>(dWhh1, HID,   32, PBdWhh1);
    pack_x_kernel<<<(TLEN * 8192 + 255) / 256, 256, 0, stream>>>(x, PAX);

    // zero initial state region (H1,H2,PAh1,PAh2 are contiguous)
    int zWords = (int)((2UL * BATCH * HID * 4 * 2 + 2UL * BATCH * HID * 2 * 2) / 4);
    zero_kernel<<<(zWords + 255) / 256, 256, 0, stream>>>((unsigned int*)(ws + oH1), zWords);

    const size_t HS = (size_t)BATCH * HID;   // per-buffer element stride
    int cur = 0;

    // ---- encoder scan ----
    for (int t = 0; t < TLEN; ++t) {
        int nxt = cur ^ 1;
        gru_cell_kernel<<<64, 128, 0, stream>>>(
            PAX + (size_t)t * 8192, PBeWih0, ebih0, 4,
            PAh1 + cur * HS, PBeWhh0, ebhh0,
            H1 + cur * HS, H1 + nxt * HS, PAh1 + nxt * HS);
        gru_cell_kernel<<<64, 128, 0, stream>>>(
            PAh1 + nxt * HS, PBeWih1, ebih1, 32,
            PAh2 + cur * HS, PBeWhh1, ebhh1,
            H2 + cur * HS, H2 + nxt * HS, PAh2 + nxt * HS);
        cur = nxt;
    }

    // ---- decoder step (dec_in == x[:, -1, :] == PAX slice t=511) ----
    int nxt = cur ^ 1;
    gru_cell_kernel<<<64, 128, 0, stream>>>(
        PAX + (size_t)(TLEN - 1) * 8192, PBdWih0, dbih0, 4,
        PAh1 + cur * HS, PBdWhh0, dbhh0,
        H1 + cur * HS, H1 + nxt * HS, PAh1 + nxt * HS);
    gru_cell_kernel<<<64, 128, 0, stream>>>(
        PAh1 + nxt * HS, PBdWih1, dbih1, 32,
        PAh2 + cur * HS, PBdWhh1, dbhh1,
        H2 + cur * HS, H2 + nxt * HS, PAh2 + nxt * HS);

    // ---- FC head ----
    fc_head_kernel<<<1, 256, 0, stream>>>(H2 + nxt * HS, fcW, fcb, outW, outb, out);
}
